// DeepFusionCrossMamba_10694468567333
// MI455X (gfx1250) — matrix operand order repaired and checked by reading, not compiled, not run on hardware
//
#include <hip/hip_runtime.h>
#include <hip/hip_bf16.h>
#include <math.h>

// ---------------- problem constants ----------------
#define T_SEQ   1024
#define D_MODEL 256
#define DI_     512
#define DS_     16
#define DTR_    16
#define NROWS   4096   // B*T

typedef __bf16 bf16_t;
typedef bf16_t v16bf __attribute__((ext_vector_type(16)));
typedef bf16_t bf8v  __attribute__((ext_vector_type(8)));
typedef bf16_t bf4v  __attribute__((ext_vector_type(4)));
typedef float  v8f   __attribute__((ext_vector_type(8)));
typedef float  f4    __attribute__((ext_vector_type(4)));

__device__ __forceinline__ float sigmoidf_(float x) { return 1.f / (1.f + __expf(-x)); }
__device__ __forceinline__ float siluf_(float x)    { return x * sigmoidf_(x); }

// ---------------- WMMA fragment helpers (v_wmma_f32_16x16x32_bf16), bf16 sources ------------
// A is 16x32 (MxK) row-major bf16. lane = 16*half + m holds K = half*8+{0..7} and +16.
__device__ __forceinline__ v16bf load_A_bf(const bf16_t* __restrict__ X, int ldx,
                                           int srcrow, int kb, int half, bool ok) {
  if (ok) {
    const bf8v* q = (const bf8v*)(X + (size_t)srcrow * ldx + kb + half * 8);
    bf8v lo = q[0], hi = q[2];   // +16 elements
    return __builtin_shufflevector(lo, hi, 0, 1, 2, 3, 4, 5, 6, 7,
                                            8, 9, 10, 11, 12, 13, 14, 15);
  }
  v16bf a;
#pragma unroll
  for (int i = 0; i < 16; ++i) a[i] = (bf16_t)0.f;
  return a;
}

// B is 32x16 (KxN); weight W is (N,K) bf16: lane = 16*kh + n holds K = kh*8+{0..7} and +16.
__device__ __forceinline__ v16bf load_B_bf(const bf16_t* __restrict__ W, int ldw,
                                           int colbase, int kb, int lane) {
  const int n = lane & 15, kh = lane >> 4;
  const bf8v* q = (const bf8v*)(W + (size_t)(colbase + n) * ldw + kb + kh * 8);
  bf8v lo = q[0], hi = q[2];
  return __builtin_shufflevector(lo, hi, 0, 1, 2, 3, 4, 5, 6, 7,
                                          8, 9, 10, 11, 12, 13, 14, 15);
}

__device__ __forceinline__ v8f wmma_bf16(v16bf a, v16bf b, v8f c) {
  return __builtin_amdgcn_wmma_f32_16x16x32_bf16(false, a, false, b, (short)0, c, false, false);
}

// ---------------- row maps (A-side indexing) ----------------
struct MapIdent {
  __device__ int operator()(int r, bool& ok) const { ok = true; return r; }
};
struct MapRev {  // reverse time within each batch (backward Mamba direction)
  __device__ int operator()(int r, bool& ok) const {
    ok = true;
    return (r & ~(T_SEQ - 1)) + (T_SEQ - 1 - (r & (T_SEQ - 1)));
  }
};

// ---------------- epilogues ----------------
struct EpiGateA2V {  // v_ref(bf16) = video * sigmoid(audio@Wt + b)
  const float* bias; const float* video; bf16_t* vref;
  __device__ void operator()(float v, int row, int col) const {
    float g = sigmoidf_(v + bias[col]);
    vref[(size_t)row * D_MODEL + col] = (bf16_t)(video[(size_t)row * D_MODEL + col] * g);
  }
};
struct EpiV2AAdd {  // x = audio + sigmoid(v_ref@Wt + b) * dlt
  const float* bias; const float* audio; const float* dlt; float* x;
  __device__ void operator()(float v, int row, int col) const {
    float g = sigmoidf_(v + bias[col]);
    x[(size_t)row * D_MODEL + col] =
        audio[(size_t)row * D_MODEL + col] + g * dlt[(size_t)row * D_MODEL + col];
  }
};
struct EpiStore {   // plain fp32 store
  float* Y; int ld;
  __device__ void operator()(float v, int row, int col) const {
    Y[(size_t)row * ld + col] = v;
  }
};
struct EpiAccumMap { // x[map(row)] += v  (out_proj; rev folds the bwd un-reverse)
  float* Y; int ld; int rev;
  __device__ void operator()(float v, int row, int col) const {
    int t = row & (T_SEQ - 1);
    int rr = rev ? (row & ~(T_SEQ - 1)) + (T_SEQ - 1 - t) : row;
    Y[(size_t)rr * ld + col] += v;
  }
};
struct EpiConvGelu { // dlt = gelu( v * bn_gamma/sqrt(1+1e-5) + bn_beta )
  const float* gamma; const float* beta; float* dlt;
  __device__ void operator()(float v, int row, int col) const {
    float s = v * gamma[col] * rsqrtf(1.f + 1e-5f) + beta[col];
    float g = 0.5f * s * (1.f + erff(s * 0.70710678118654752f));
    dlt[(size_t)row * D_MODEL + col] = g;
  }
};

// ---------------- generic WMMA GEMM: out[m,n] = epi( sum_k X[map(m),k]*W[n,k] ) ----------------
// block = 128 (4 waves); wave w owns MT consecutive M-tiles and NT N-tiles (32x64 patch
// for MT=2,NT=4): per K-step 4 A-loads + 8 B-loads feed 8 WMMAs (1.5 b128 per WMMA).
template <int MT, int NT, class Map, class Epi>
__global__ __launch_bounds__(128) void gemm_wmma(const bf16_t* __restrict__ X, int ldx,
                                                 const bf16_t* __restrict__ W, int ldw,
                                                 int N, int K, Map map, Epi epi) {
  const int lane = threadIdx.x & 31;
  const int wave = threadIdx.x >> 5;
  const int mtile0 = (blockIdx.x * 4 + wave) * MT;
  const int ntile0 = blockIdx.y * NT;
  const int m = lane & 15, half = lane >> 4;

  bool ok[MT];
  int srcrow[MT];
#pragma unroll
  for (int i = 0; i < MT; ++i) srcrow[i] = map((mtile0 + i) * 16 + m, ok[i]);

  v8f acc[MT][NT] = {};
  for (int kb = 0; kb < K; kb += 32) {
    v16bf a[MT];
#pragma unroll
    for (int i = 0; i < MT; ++i) a[i] = load_A_bf(X, ldx, srcrow[i], kb, half, ok[i]);
#pragma unroll
    for (int j = 0; j < NT; ++j) {
      if ((ntile0 + j) * 16 < N) {
        v16bf b = load_B_bf(W, ldw, (ntile0 + j) * 16, kb, lane);
#pragma unroll
        for (int i = 0; i < MT; ++i) acc[i][j] = wmma_bf16(a[i], b, acc[i][j]);
      }
    }
  }
  const int n = lane & 15;
#pragma unroll
  for (int i = 0; i < MT; ++i)
#pragma unroll
    for (int j = 0; j < NT; ++j) {
      if ((ntile0 + j) * 16 < N) {
#pragma unroll
        for (int r = 0; r < 8; ++r) {
          epi(acc[i][j][r], (mtile0 + i) * 16 + half * 8 + r, (ntile0 + j) * 16 + n);
        }
      }
    }
}

// ---------------- dense temporal conv front (3 shifted GEMMs fused, k-major bf16 weights) ----
__global__ __launch_bounds__(128) void conv_front(const bf16_t* __restrict__ vref,
                                                  const bf16_t* __restrict__ pwT, // [3][D][D]
                                                  const float* __restrict__ gamma,
                                                  const float* __restrict__ beta,
                                                  float* __restrict__ dlt) {
  const int lane = threadIdx.x & 31;
  const int wave = threadIdx.x >> 5;
  const int mtile = blockIdx.x * 4 + wave;
  const int ntile0 = blockIdx.y * 4;
  const int rowbase = mtile * 16;
  const int m = lane & 15, half = lane >> 4;

  v8f acc[4] = {};
  for (int sh = -1; sh <= 1; ++sh) {
    int trow = rowbase + m;
    int t = (trow & (T_SEQ - 1)) + sh;
    bool ok = (t >= 0) && (t < T_SEQ);
    int srcrow = (trow & ~(T_SEQ - 1)) + t;
    const bf16_t* Wk = pwT + (size_t)(sh + 1) * D_MODEL * D_MODEL;
    for (int kb = 0; kb < D_MODEL; kb += 32) {
      v16bf a = load_A_bf(vref, D_MODEL, srcrow, kb, half, ok);
#pragma unroll
      for (int j = 0; j < 4; ++j) {
        v16bf b = load_B_bf(Wk, D_MODEL, (ntile0 + j) * 16, kb, lane);
        acc[j] = wmma_bf16(a, b, acc[j]);
      }
    }
  }
  EpiConvGelu epi{gamma, beta, dlt};
  const int n = lane & 15;
#pragma unroll
  for (int j = 0; j < 4; ++j)
#pragma unroll
    for (int r = 0; r < 8; ++r)
      epi(acc[j][r], rowbase + half * 8 + r, (ntile0 + j) * 16 + n);
}

// ---------------- prep: fp32 -> bf16 conversion (4 elems/thread) ----------------
__global__ __launch_bounds__(256) void cvt_f32_bf16(const float* __restrict__ s,
                                                    bf16_t* __restrict__ d, int n4) {
  int i = blockIdx.x * 256 + threadIdx.x;
  if (i < n4) {
    f4 v = ((const f4*)s)[i];
    bf4v o;
    o[0] = (bf16_t)v[0]; o[1] = (bf16_t)v[1]; o[2] = (bf16_t)v[2]; o[3] = (bf16_t)v[3];
    ((bf4v*)d)[i] = o;
  }
}

// ---------------- prep: repack conv weight (D,D,3) -> [3][D][D] bf16 ----------------
__global__ __launch_bounds__(256) void repack_conv(const float* __restrict__ pw,
                                                   bf16_t* __restrict__ pwT) {
  int idx = blockIdx.x * 256 + threadIdx.x;   // over e*d (65536)
  int e = idx >> 8, d = idx & 255;
#pragma unroll
  for (int k = 0; k < 3; ++k)
    pwT[(size_t)k * D_MODEL * D_MODEL + idx] = (bf16_t)pw[((size_t)e * D_MODEL + d) * 3 + k];
}

// ---------------- rmsnorm (per row of 256) -> bf16 h ----------------
__global__ __launch_bounds__(256) void rmsnorm_kernel(const float* __restrict__ x,
                                                      const float* __restrict__ w,
                                                      bf16_t* __restrict__ h) {
  __shared__ float sm[256];
  const int r = blockIdx.x, tid = threadIdx.x;
  float v = x[(size_t)r * D_MODEL + tid];
  sm[tid] = v * v; __syncthreads();
  for (int o = 128; o > 0; o >>= 1) { if (tid < o) sm[tid] += sm[tid + o]; __syncthreads(); }
  float inv = rsqrtf(sm[0] / (float)D_MODEL + 1e-5f);
  h[(size_t)r * D_MODEL + tid] = (bf16_t)(v * inv * w[tid]);
}

// ---------------- causal depthwise conv (DCONV=4) + silu; fp32 + bf16 outputs ----------------
__global__ __launch_bounds__(512) void dwconv_kernel(const float* __restrict__ xz,
                                                     const float* __restrict__ cw,  // (DI,4)
                                                     const float* __restrict__ cb,
                                                     float* __restrict__ xc,
                                                     bf16_t* __restrict__ xc_bf) {
  const int r = blockIdx.x, d = threadIdx.x;
  const int t = r & (T_SEQ - 1);
  const int b0 = r & ~(T_SEQ - 1);
  float acc = cb[d];
#pragma unroll
  for (int j = 0; j < 4; ++j) {
    int tt = t - 3 + j;
    if (tt >= 0) acc += xz[(size_t)(b0 + tt) * (2 * DI_) + d] * cw[d * 4 + j];
  }
  float s = siluf_(acc);
  xc[(size_t)r * DI_ + d] = s;
  xc_bf[(size_t)r * DI_ + d] = (bf16_t)s;
}

// ---------------- delta = softplus(dt @ dtw^T + dtb), K=16 (VALU, LDS-shared dt) ----------------
__global__ __launch_bounds__(512) void delta_kernel(const float* __restrict__ xdbl,
                                                    const float* __restrict__ dtw,  // (DI,16)
                                                    const float* __restrict__ dtb,
                                                    float* __restrict__ delta) {
  __shared__ float sdt[DTR_];
  const int r = blockIdx.x, d = threadIdx.x;
  if (d < DTR_) sdt[d] = xdbl[(size_t)r * 48 + d];
  __syncthreads();
  float acc = dtb[d];
#pragma unroll
  for (int k = 0; k < DTR_; ++k) acc += sdt[k] * dtw[d * DTR_ + k];
  float sp = (acc > 20.f) ? acc : log1pf(__expf(acc));
  delta[(size_t)r * DI_ + d] = sp;
}

// ---------------- selective-scan: h = exp(delta*A)*h + delta*xc*B; y = h.C (bf16 out) --------
__global__ __launch_bounds__(128) void scan_kernel(const float* __restrict__ delta,
                                                   const float* __restrict__ xc,
                                                   const float* __restrict__ xdbl,
                                                   const float* __restrict__ xz,
                                                   const float* __restrict__ Alog,
                                                   const float* __restrict__ Dsk,
                                                   bf16_t* __restrict__ y) {
  const int b = blockIdx.x;
  const int d = blockIdx.y * 128 + threadIdx.x;
  __shared__ float sB[DS_], sC[DS_];
  float A[DS_], h[DS_];
#pragma unroll
  for (int s = 0; s < DS_; ++s) { A[s] = -__expf(Alog[d * DS_ + s]); h[s] = 0.f; }
  const float dsk = Dsk[d];
  for (int t = 0; t < T_SEQ; ++t) {
    size_t r = (size_t)b * T_SEQ + t;
    if (threadIdx.x < DS_)            sB[threadIdx.x]        = xdbl[r * 48 + 16 + threadIdx.x];
    else if (threadIdx.x < 2 * DS_)   sC[threadIdx.x - DS_]  = xdbl[r * 48 + 32 + (threadIdx.x - DS_)];
    __syncthreads();
    float dl  = delta[r * DI_ + d];
    float xcv = xc[r * DI_ + d];
    float du  = dl * xcv;
    float yv  = 0.f;
#pragma unroll
    for (int s = 0; s < DS_; ++s) {
      h[s] = __expf(dl * A[s]) * h[s] + du * sB[s];
      yv += h[s] * sC[s];
    }
    yv += dsk * xcv;
    float zv = xz[r * (2 * DI_) + DI_ + d];
    yv *= siluf_(zv);
    y[r * DI_ + d] = (bf16_t)yv;
    __syncthreads();
  }
}

// ---------------- final layernorm: out = gamma*(s-mean)/sqrt(var+eps)+beta, s=x+audio --------
__global__ __launch_bounds__(256) void final_ln(const float* __restrict__ x,
                                                const float* __restrict__ audio,
                                                const float* __restrict__ gamma,
                                                const float* __restrict__ beta,
                                                float* __restrict__ out) {
  __shared__ float sm[256];
  const int r = blockIdx.x, tid = threadIdx.x;
  float s = x[(size_t)r * D_MODEL + tid] + audio[(size_t)r * D_MODEL + tid];
  sm[tid] = s; __syncthreads();
  for (int o = 128; o > 0; o >>= 1) { if (tid < o) sm[tid] += sm[tid + o]; __syncthreads(); }
  float mean = sm[0] / (float)D_MODEL; __syncthreads();
  float dv = s - mean;
  sm[tid] = dv * dv; __syncthreads();
  for (int o = 128; o > 0; o >>= 1) { if (tid < o) sm[tid] += sm[tid + o]; __syncthreads(); }
  float var = sm[0] / (float)D_MODEL;
  out[(size_t)r * D_MODEL + tid] = gamma[tid] * dv * rsqrtf(var + 1e-8f) + beta[tid];
}

// ---------------- host orchestration ----------------
extern "C" void kernel_launch(void* const* d_in, const int* in_sizes, int n_in,
                              void* d_out, int out_size, void* d_ws, size_t ws_size,
                              hipStream_t stream) {
  (void)in_sizes; (void)n_in; (void)out_size; (void)ws_size;
  const float* audio      = (const float*)d_in[0];
  const float* video      = (const float*)d_in[1];
  const float* gate_a2v_w = (const float*)d_in[2];
  const float* gate_a2v_b = (const float*)d_in[3];
  const float* gate_v2a_w = (const float*)d_in[4];
  const float* gate_v2a_b = (const float*)d_in[5];
  const float* proj_w     = (const float*)d_in[6];
  const float* bn_gamma   = (const float*)d_in[7];
  const float* bn_beta    = (const float*)d_in[8];
  const float* rms_w      = (const float*)d_in[9];
  const float* in_proj_w  = (const float*)d_in[10];
  const float* conv_w     = (const float*)d_in[11];
  const float* conv_b     = (const float*)d_in[12];
  const float* x_proj_w   = (const float*)d_in[13];
  const float* dt_w       = (const float*)d_in[14];
  const float* dt_b       = (const float*)d_in[15];
  const float* A_log      = (const float*)d_in[16];
  const float* D_skip     = (const float*)d_in[17];
  const float* out_w      = (const float*)d_in[18];
  const float* cln_gamma  = (const float*)d_in[19];
  const float* cln_beta   = (const float*)d_in[20];
  float* out = (float*)d_out;

  // ---- workspace carve-up: fp32 region, then bf16 region ----
  const size_t nBT_D  = (size_t)NROWS * D_MODEL;   // 1,048,576
  const size_t nBT_DI = (size_t)NROWS * DI_;       // 2,097,152
  float* ws    = (float*)d_ws;
  float* dlt   = ws;
  float* x     = dlt   + nBT_D;
  float* xz    = x     + nBT_D;                    // NROWS*1024
  float* xc    = xz    + (size_t)NROWS * 2 * DI_;
  float* xdbl  = xc    + nBT_DI;                   // NROWS*48
  float* delta = xdbl  + (size_t)NROWS * 48;
  bf16_t* bfp  = (bf16_t*)(delta + nBT_DI);
  bf16_t* audio_bf = bfp;            bfp += nBT_D;
  bf16_t* vref_bf  = bfp;            bfp += nBT_D;
  bf16_t* h_bf     = bfp;            bfp += nBT_D;
  bf16_t* xc_bf    = bfp;            bfp += nBT_DI;
  bf16_t* y_bf     = bfp;            bfp += nBT_DI;
  bf16_t* w_a2v    = bfp;            bfp += (size_t)D_MODEL * D_MODEL;
  bf16_t* w_v2a    = bfp;            bfp += (size_t)D_MODEL * D_MODEL;
  bf16_t* pwT      = bfp;            bfp += (size_t)3 * D_MODEL * D_MODEL;
  bf16_t* w_in     = bfp;            bfp += (size_t)4 * 2 * DI_ * D_MODEL;
  bf16_t* w_xp     = bfp;            bfp += (size_t)4 * 48 * DI_;
  bf16_t* w_out    = bfp;            bfp += (size_t)4 * D_MODEL * DI_;

  // ---- prep: weight/activation conversion to bf16 (once per launch) ----
  auto cvt = [&](const float* s, bf16_t* d, size_t n) {
    int n4 = (int)(n / 4);
    cvt_f32_bf16<<<(n4 + 255) / 256, 256, 0, stream>>>(s, d, n4);
  };
  cvt(audio, audio_bf, nBT_D);
  cvt(gate_a2v_w, w_a2v, (size_t)D_MODEL * D_MODEL);
  cvt(gate_v2a_w, w_v2a, (size_t)D_MODEL * D_MODEL);
  cvt(in_proj_w,  w_in,  (size_t)4 * 2 * DI_ * D_MODEL);
  cvt(x_proj_w,   w_xp,  (size_t)4 * 48 * DI_);
  cvt(out_w,      w_out, (size_t)4 * D_MODEL * DI_);
  repack_conv<<<(D_MODEL * D_MODEL) / 256, 256, 0, stream>>>(proj_w, pwT);

  const dim3 blk(128);
  const dim3 gD(NROWS / 128, D_MODEL / 64);   // MT=2: 32x64 per wave, N=256
  const dim3 gIn(NROWS / 128, (2 * DI_) / 64);// N=1024
  const dim3 gXp(NROWS / 128, 1);             // N=48 (guarded)
  const dim3 gCv(NROWS / 64, D_MODEL / 64);   // conv_front keeps MT=1 tiles

  // ---- front-end ----
  gemm_wmma<2, 4, MapIdent, EpiGateA2V><<<gD, blk, 0, stream>>>(
      audio_bf, D_MODEL, w_a2v, D_MODEL, D_MODEL, D_MODEL,
      MapIdent{}, EpiGateA2V{gate_a2v_b, video, vref_bf});
  conv_front<<<gCv, blk, 0, stream>>>(vref_bf, pwT, bn_gamma, bn_beta, dlt);
  gemm_wmma<2, 4, MapIdent, EpiV2AAdd><<<gD, blk, 0, stream>>>(
      vref_bf, D_MODEL, w_v2a, D_MODEL, D_MODEL, D_MODEL,
      MapIdent{}, EpiV2AAdd{gate_v2a_b, audio, dlt, x});

  // ---- bidirectional Mamba layers ----
  for (int i = 0; i < 2; ++i) {
    rmsnorm_kernel<<<NROWS, 256, 0, stream>>>(x, rms_w + (size_t)i * D_MODEL, h_bf);
    for (int dir = 0; dir < 2; ++dir) {
      const int pi = i * 2 + dir;
      const bf16_t* inw = w_in + (size_t)pi * 2 * DI_ * D_MODEL;
      if (dir == 0)
        gemm_wmma<2, 4, MapIdent, EpiStore><<<gIn, blk, 0, stream>>>(
            h_bf, D_MODEL, inw, D_MODEL, 2 * DI_, D_MODEL,
            MapIdent{}, EpiStore{xz, 2 * DI_});
      else
        gemm_wmma<2, 4, MapRev, EpiStore><<<gIn, blk, 0, stream>>>(
            h_bf, D_MODEL, inw, D_MODEL, 2 * DI_, D_MODEL,
            MapRev{}, EpiStore{xz, 2 * DI_});

      dwconv_kernel<<<NROWS, DI_, 0, stream>>>(
          xz, conv_w + (size_t)pi * DI_ * 4, conv_b + (size_t)pi * DI_, xc, xc_bf);

      gemm_wmma<2, 4, MapIdent, EpiStore><<<gXp, blk, 0, stream>>>(
          xc_bf, DI_, w_xp + (size_t)pi * 48 * DI_, DI_, 48, DI_,
          MapIdent{}, EpiStore{xdbl, 48});

      delta_kernel<<<NROWS, DI_, 0, stream>>>(
          xdbl, dt_w + (size_t)pi * DI_ * DTR_, dt_b + (size_t)pi * DI_, delta);

      scan_kernel<<<dim3(4, DI_ / 128), 128, 0, stream>>>(
          delta, xc, xdbl, xz,
          A_log + (size_t)pi * DI_ * DS_, D_skip + (size_t)pi * DI_, y_bf);

      gemm_wmma<2, 4, MapIdent, EpiAccumMap><<<gD, blk, 0, stream>>>(
          y_bf, DI_, w_out + (size_t)pi * D_MODEL * DI_, DI_, D_MODEL, DI_,
          MapIdent{}, EpiAccumMap{x, D_MODEL, dir});
    }
  }

  final_ln<<<NROWS, 256, 0, stream>>>(x, audio, cln_gamma, cln_beta, out);
}
